// LSTM_46514495816262
// MI455X (gfx1250) — matrix operand order repaired and checked by reading, not compiled, not run on hardware
//
#include <hip/hip_runtime.h>
#include <hip/hip_bf16.h>

// Problem constants (match reference)
#define TT 2048   // seq len
#define BB 32     // batch
#define HH 512    // hidden
#define GG (4 * HH)
#define LL 2
#define NWG 8     // persistent grid: 8 blocks x 256 threads = 64 waves
#define CT 512    // time chunk for gates_x precompute

typedef __attribute__((ext_vector_type(16))) __bf16 v16bf;
typedef __attribute__((ext_vector_type(8)))  __bf16 v8bf;
typedef __attribute__((ext_vector_type(8)))  float  v8f;

// ---------------------------------------------------------------------------
// D = A(16x32 bf16) * B(32x16 bf16) + C(16x16 f32)
// ---------------------------------------------------------------------------
__device__ __forceinline__ v8f wmma_bf16(v16bf a, v16bf b, v8f c) {
  return __builtin_amdgcn_wmma_f32_16x16x32_bf16(
      /*neg_a=*/false, a, /*neg_b=*/false, b,
      /*c_mod=*/(short)0, c, /*reuse_a=*/false, /*reuse_b=*/false);
}

__device__ __forceinline__ float sigmoidf_fast(float x) {
  return 1.0f / (1.0f + __expf(-x));
}
__device__ __forceinline__ float tanhf_fast(float x) {
  float e = __expf(2.0f * x);
  return (e - 1.0f) / (e + 1.0f);
}

// A-fragment (16x32, bf16 source): lane holds row M=lane&15,
// K = base..base+15 contiguous with base = (lane>>4)*16.
__device__ __forceinline__ v16bf load_a_bf(const __bf16* __restrict__ row,
                                           int k0, int abase) {
  v8bf lo = *(const v8bf*)(row + k0 + 2 * abase);
  v8bf hi = *(const v8bf*)(row + k0 + 2 * abase + 8);
  v16bf a;
#pragma unroll
  for (int i = 0; i < 8; ++i) { a[i] = lo[i]; a[i + 8] = hi[i]; }
  return a;
}

// A-fragment from an LDS row (same packing; compiler infers addrspace(3)
// after inlining -> ds_load_b128).
__device__ __forceinline__ v16bf load_a_lds(const __bf16* row, int k0,
                                            int abase) {
  v8bf lo = *(const v8bf*)(row + k0 + 2 * abase);
  v8bf hi = *(const v8bf*)(row + k0 + 2 * abase + 8);
  v16bf a;
#pragma unroll
  for (int i = 0; i < 8; ++i) { a[i] = lo[i]; a[i + 8] = hi[i]; }
  return a;
}

// Same A-fragment but converting from fp32 on the fly.
__device__ __forceinline__ v16bf load_a_f32(const float* __restrict__ row,
                                            int k0, int abase) {
  v8f lo = *(const v8f*)(row + k0 + 2 * abase);
  v8f hi = *(const v8f*)(row + k0 + 2 * abase + 8);
  v16bf a;
#pragma unroll
  for (int i = 0; i < 8; ++i) { a[i] = (__bf16)lo[i]; a[i + 8] = (__bf16)hi[i]; }
  return a;
}

// Device-wide barrier (8 workgroups, all resident).
__device__ __forceinline__ void grid_sync(int* bar, int* phase) {
  __syncthreads();
  (*phase)++;
  if (threadIdx.x == 0) {
    __threadfence();
    atomicAdd(bar, 1);
    const int target = NWG * (*phase);
    while (__hip_atomic_load(bar, __ATOMIC_RELAXED, __HIP_MEMORY_SCOPE_AGENT) <
           target) {
      __builtin_amdgcn_s_sleep(1);
    }
    __threadfence();
  }
  __syncthreads();
}

// fp32 -> bf16 weight conversion (both matrices at once)
__global__ void cvt_weights(const float* __restrict__ wih,
                            const float* __restrict__ whh,
                            __bf16* __restrict__ wih_b,
                            __bf16* __restrict__ whh_b, int n) {
  for (int i = blockIdx.x * blockDim.x + threadIdx.x; i < n;
       i += gridDim.x * blockDim.x) {
    wih_b[i] = (__bf16)wih[i];
    whh_b[i] = (__bf16)whh[i];
  }
}

// ---------------------------------------------------------------------------
// Parallel gates_x GEMM: gx[t-t0, g, b] = sum_k x[t,b,k] * w[g,k]  (bf16 out,
// TRANSPOSED [CT,G,B] layout so the scan kernel seeds accumulators with one
// contiguous v8bf load per gate). Each wave: one (m-tile, n-block) task;
// m-tile = 16 rows of (t, b-half), n-block = 64 gate columns (4 WMMA tiles).
// ---------------------------------------------------------------------------
template <bool XF32>
__global__ void __launch_bounds__(256) gates_gemm(
    const float*  __restrict__ xf,   // [T,B,H] fp32 (layer 0)
    const __bf16* __restrict__ xb,   // [T,B,H] bf16 (layer 1)
    const __bf16* __restrict__ w,    // [G,H] bf16
    __bf16* __restrict__ gx,         // [CT,G,B] bf16
    int t0, int ct) {
  const int tid   = threadIdx.x;
  const int lane  = tid & 31;
  const int wtask = blockIdx.x * 8 + (tid >> 5);  // 0 .. 2*ct*32-1
  const int mt    = wtask >> 5;                   // 0 .. 2*ct-1
  const int nb    = wtask & 31;                   // 0 .. 31
  const int t     = t0 + (mt >> 1);
  const int b0    = (mt & 1) * 16;
  const int n0    = nb * 64;
  const int nhalf = lane >> 4;
  const int nl    = lane & 15;
  const int abase = nhalf * 8;

  const float*  xrow_f =
      XF32 ? xf + ((size_t)t * BB + b0 + nl) * HH : nullptr;
  const __bf16* xrow_b =
      XF32 ? nullptr : xb + ((size_t)t * BB + b0 + nl) * HH;

  const __bf16* wr[4];
#pragma unroll
  for (int j = 0; j < 4; ++j)
    wr[j] = w + (size_t)(n0 + j * 16 + nl) * HH + (size_t)nhalf * 16;

  v8f acc[4];
#pragma unroll
  for (int j = 0; j < 4; ++j) acc[j] = (v8f){0, 0, 0, 0, 0, 0, 0, 0};

#pragma unroll 2
  for (int ks = 0; ks < HH / 32; ++ks) {
    const int k0 = ks * 32;
    const v16bf a = XF32 ? load_a_f32(xrow_f, k0, abase)
                         : load_a_bf(xrow_b, k0, abase);
    const v16bf bfr0 = *(const v16bf*)(wr[0] + k0);
    const v16bf bfr1 = *(const v16bf*)(wr[1] + k0);
    const v16bf bfr2 = *(const v16bf*)(wr[2] + k0);
    const v16bf bfr3 = *(const v16bf*)(wr[3] + k0);
    acc[0] = wmma_bf16(a, bfr0, acc[0]);
    acc[1] = wmma_bf16(a, bfr1, acc[1]);
    acc[2] = wmma_bf16(a, bfr2, acc[2]);
    acc[3] = wmma_bf16(a, bfr3, acc[3]);
  }

  // store transposed: 8 consecutive batch rows per lane -> one v8bf store
#pragma unroll
  for (int j = 0; j < 4; ++j) {
    v8bf o;
#pragma unroll
    for (int v = 0; v < 8; ++v) o[v] = (__bf16)acc[j][v];
    const size_t base =
        ((size_t)(t - t0) * GG + n0 + j * 16 + nl) * BB + b0 + 8 * nhalf;
    *(v8bf*)(gx + base) = o;
  }
}

// ---------------------------------------------------------------------------
// FAST PATH sequential scan over one time chunk. Accumulators seeded from the
// precomputed gates_x; only the h @ w_hh^T WMMA chain runs per step. All 8
// waves of a block share batch rows [16r,16r+16), so the block stages its
// 16KB h-slice in LDS once per step and A-fragments come from ds_load.
// ---------------------------------------------------------------------------
template <int LAYER>
__global__ void __launch_bounds__(256) lstm_seq(
    const __bf16* __restrict__ gx,    // [CT,G,B] this chunk
    const __bf16* __restrict__ whh,   // [G,H]
    const float*  __restrict__ bih,   // [G]
    const float*  __restrict__ bhh,   // [G]
    const float*  __restrict__ h0_or_null,  // init hbuf when non-null
    const float*  __restrict__ cin,   // c0 (chunk 0) or cbuf
    float*  __restrict__ cbuf,        // c persisted at chunk end
    __bf16* __restrict__ hbuf,        // [2,B,H] bf16 ping-pong
    __bf16* __restrict__ xout_bf,     // LAYER==0: next layer input [T,B,H]
    float*  __restrict__ out_f32,     // LAYER==1: d_out ys [T,B,H]
    float*  __restrict__ hT_out,      // [B,H]
    float*  __restrict__ cT_out,      // [B,H]
    int t0, int ct, int* bar) {
  __shared__ __bf16 hsh[16 * HH];     // this block's 16 batch rows of h

  const int tid   = threadIdx.x;
  const int lane  = tid & 31;
  const int wid   = blockIdx.x * 8 + (tid >> 5);  // 0..63
  const int r     = wid >> 5;                     // same for all waves in blk
  const int jt    = wid & 31;
  const int nhalf = lane >> 4;
  const int nl    = lane & 15;
  const int abase = nhalf * 8;

  int phase = 0;

  if (h0_or_null) {
    for (int i = blockIdx.x * blockDim.x + tid; i < BB * HH;
         i += gridDim.x * blockDim.x)
      hbuf[i] = (__bf16)h0_or_null[i];
  }
  grid_sync(bar, &phase);

  const int brow0 = 16 * r + 8 * nhalf;
  const int jcol  = jt * 16 + nl;

  const float bias_i = bih[0 * HH + jcol] + bhh[0 * HH + jcol];
  const float bias_f = bih[1 * HH + jcol] + bhh[1 * HH + jcol];
  const float bias_g = bih[2 * HH + jcol] + bhh[2 * HH + jcol];
  const float bias_o = bih[3 * HH + jcol] + bhh[3 * HH + jcol];

  v8f cst;
#pragma unroll
  for (int v = 0; v < 8; ++v) cst[v] = cin[(size_t)(brow0 + v) * HH + jcol];

  const __bf16* wh[4];
#pragma unroll
  for (int g = 0; g < 4; ++g)
    wh[g] = whh + (size_t)(g * HH + jcol) * HH + (size_t)nhalf * 16;

  for (int t = t0; t < t0 + ct; ++t) {
    const int buf  = t & 1;
    const int nbuf = buf ^ 1;

    // stage this block's h rows into LDS (uniform loop -> no divergence)
    {
      const v8bf* src =
          (const v8bf*)(hbuf + (size_t)buf * BB * HH + (size_t)16 * r * HH);
      v8bf* dst = (v8bf*)hsh;
#pragma unroll
      for (int i = 0; i < (16 * HH / 8) / 256; ++i)
        dst[tid + i * 256] = src[tid + i * 256];
    }
    __syncthreads();

    // seed accumulators with precomputed x-side gates (+ prefetch next t)
    const size_t gxt = (size_t)(t - t0) * GG;
    v8f acc[4];
#pragma unroll
    for (int g = 0; g < 4; ++g) {
      const v8bf gv = *(const v8bf*)(gx + (gxt + g * HH + jcol) * BB + brow0);
#pragma unroll
      for (int v = 0; v < 8; ++v) acc[g][v] = (float)gv[v];
    }
    if (t + 1 < t0 + ct)
      __builtin_prefetch(gx + (gxt + GG) * BB + jcol * BB + brow0, 0, 1);

    const __bf16* hrow = hsh + (size_t)nl * HH;  // A row M = nl, from LDS

#pragma unroll 4
    for (int ks = 0; ks < HH / 32; ++ks) {
      const int k0 = ks * 32;
      const v16bf ah  = load_a_lds(hrow, k0, abase);
      const v16bf bh0 = *(const v16bf*)(wh[0] + k0);
      const v16bf bh1 = *(const v16bf*)(wh[1] + k0);
      const v16bf bh2 = *(const v16bf*)(wh[2] + k0);
      const v16bf bh3 = *(const v16bf*)(wh[3] + k0);
      acc[0] = wmma_bf16(ah, bh0, acc[0]);
      acc[1] = wmma_bf16(ah, bh1, acc[1]);
      acc[2] = wmma_bf16(ah, bh2, acc[2]);
      acc[3] = wmma_bf16(ah, bh3, acc[3]);
    }

#pragma unroll
    for (int v = 0; v < 8; ++v) {
      const float iv = sigmoidf_fast(acc[0][v] + bias_i);
      const float fv = sigmoidf_fast(acc[1][v] + bias_f);
      const float gv = tanhf_fast(acc[2][v] + bias_g);
      const float ov = sigmoidf_fast(acc[3][v] + bias_o);
      cst[v] = fv * cst[v] + iv * gv;
      const float hv = ov * tanhf_fast(cst[v]);

      const size_t off = (size_t)(brow0 + v) * HH + jcol;
      hbuf[(size_t)nbuf * BB * HH + off] = (__bf16)hv;
      if (LAYER == 0) xout_bf[(size_t)t * BB * HH + off] = (__bf16)hv;
      if (LAYER == 1) out_f32[(size_t)t * BB * HH + off] = hv;
      if (t == TT - 1) { hT_out[off] = hv; cT_out[off] = cst[v]; }
    }

    grid_sync(bar, &phase);  // also fences hsh reuse for next iteration
  }

  // persist cell state for the next chunk
#pragma unroll
  for (int v = 0; v < 8; ++v)
    cbuf[(size_t)(brow0 + v) * HH + jcol] = cst[v];
}

// ---------------------------------------------------------------------------
// FALLBACK fused persistent kernel: both GEMMs per step. Used when ws_size
// can't hold the gates_x chunk buffer.
// ---------------------------------------------------------------------------
template <int LAYER>
__global__ void __launch_bounds__(256) lstm_fused(
    const float*  __restrict__ x_f32,
    const __bf16* __restrict__ x_bf,
    const __bf16* __restrict__ wih,
    const __bf16* __restrict__ whh,
    const float*  __restrict__ bih,
    const float*  __restrict__ bhh,
    const float*  __restrict__ h0,
    const float*  __restrict__ c0,
    __bf16* __restrict__ hbuf,
    __bf16* __restrict__ xout_bf,
    float*  __restrict__ out_f32,
    float*  __restrict__ hT_out,
    float*  __restrict__ cT_out,
    int* bar) {
  const int tid   = threadIdx.x;
  const int lane  = tid & 31;
  const int wid   = blockIdx.x * 8 + (tid >> 5);
  const int r     = wid >> 5;
  const int jt    = wid & 31;
  const int nhalf = lane >> 4;
  const int nl    = lane & 15;
  const int abase = nhalf * 8;

  int phase = 0;

  for (int i = blockIdx.x * blockDim.x + tid; i < BB * HH;
       i += gridDim.x * blockDim.x)
    hbuf[i] = (__bf16)h0[i];
  grid_sync(bar, &phase);

  const int brow0 = 16 * r + 8 * nhalf;
  const int jcol  = jt * 16 + nl;

  const float bias_i = bih[0 * HH + jcol] + bhh[0 * HH + jcol];
  const float bias_f = bih[1 * HH + jcol] + bhh[1 * HH + jcol];
  const float bias_g = bih[2 * HH + jcol] + bhh[2 * HH + jcol];
  const float bias_o = bih[3 * HH + jcol] + bhh[3 * HH + jcol];

  v8f cst;
#pragma unroll
  for (int v = 0; v < 8; ++v) cst[v] = c0[(size_t)(brow0 + v) * HH + jcol];

  const __bf16* wi[4];
  const __bf16* wh[4];
#pragma unroll
  for (int g = 0; g < 4; ++g) {
    const size_t row = (size_t)(g * HH + jcol) * HH + (size_t)nhalf * 16;
    wi[g] = wih + row;
    wh[g] = whh + row;
  }

  const int arow = 16 * r + nl;

  for (int t = 0; t < TT; ++t) {
    const int buf  = t & 1;
    const int nbuf = buf ^ 1;

    const float*  xrow_f =
        (LAYER == 0) ? x_f32 + ((size_t)t * BB + arow) * HH : nullptr;
    const __bf16* xrow_b =
        (LAYER == 1) ? x_bf + ((size_t)t * BB + arow) * HH : nullptr;
    const __bf16* hrow = hbuf + (size_t)buf * BB * HH + (size_t)arow * HH;

    if (t + 1 < TT) {
      if (LAYER == 0) __builtin_prefetch(xrow_f + BB * HH, 0, 1);
      else            __builtin_prefetch(xrow_b + BB * HH, 0, 1);
    }

    v8f acc0 = {0, 0, 0, 0, 0, 0, 0, 0};
    v8f acc1 = {0, 0, 0, 0, 0, 0, 0, 0};
    v8f acc2 = {0, 0, 0, 0, 0, 0, 0, 0};
    v8f acc3 = {0, 0, 0, 0, 0, 0, 0, 0};

#pragma unroll 2
    for (int ks = 0; ks < HH / 32; ++ks) {
      const int k0 = ks * 32;
      const v16bf ax = (LAYER == 0) ? load_a_f32(xrow_f, k0, abase)
                                    : load_a_bf(xrow_b, k0, abase);
      const v16bf ah  = load_a_bf(hrow, k0, abase);
      const v16bf bi0 = *(const v16bf*)(wi[0] + k0);
      const v16bf bh0 = *(const v16bf*)(wh[0] + k0);
      const v16bf bi1 = *(const v16bf*)(wi[1] + k0);
      const v16bf bh1 = *(const v16bf*)(wh[1] + k0);
      const v16bf bi2 = *(const v16bf*)(wi[2] + k0);
      const v16bf bh2 = *(const v16bf*)(wh[2] + k0);
      const v16bf bi3 = *(const v16bf*)(wi[3] + k0);
      const v16bf bh3 = *(const v16bf*)(wh[3] + k0);

      acc0 = wmma_bf16(ax, bi0, acc0);
      acc0 = wmma_bf16(ah, bh0, acc0);
      acc1 = wmma_bf16(ax, bi1, acc1);
      acc1 = wmma_bf16(ah, bh1, acc1);
      acc2 = wmma_bf16(ax, bi2, acc2);
      acc2 = wmma_bf16(ah, bh2, acc2);
      acc3 = wmma_bf16(ax, bi3, acc3);
      acc3 = wmma_bf16(ah, bh3, acc3);
    }

#pragma unroll
    for (int v = 0; v < 8; ++v) {
      const float iv = sigmoidf_fast(acc0[v] + bias_i);
      const float fv = sigmoidf_fast(acc1[v] + bias_f);
      const float gv = tanhf_fast(acc2[v] + bias_g);
      const float ov = sigmoidf_fast(acc3[v] + bias_o);
      cst[v] = fv * cst[v] + iv * gv;
      const float hv = ov * tanhf_fast(cst[v]);

      const size_t off = (size_t)(brow0 + v) * HH + jcol;
      hbuf[(size_t)nbuf * BB * HH + off] = (__bf16)hv;
      if (LAYER == 0) xout_bf[(size_t)t * BB * HH + off] = (__bf16)hv;
      if (LAYER == 1) out_f32[(size_t)t * BB * HH + off] = hv;
      if (t == TT - 1) { hT_out[off] = hv; cT_out[off] = cst[v]; }
    }

    grid_sync(bar, &phase);
  }
}

// ---------------------------------------------------------------------------
// Host-side launch.
// ---------------------------------------------------------------------------
extern "C" void kernel_launch(void* const* d_in, const int* in_sizes, int n_in,
                              void* d_out, int out_size, void* d_ws,
                              size_t ws_size, hipStream_t stream) {
  const float* x   = (const float*)d_in[0];  // [T,B,H]
  const float* wih = (const float*)d_in[1];  // [L,G,H]
  const float* whh = (const float*)d_in[2];  // [L,G,H]
  const float* bih = (const float*)d_in[3];  // [L,G]
  const float* bhh = (const float*)d_in[4];  // [L,G]
  const float* h0  = (const float*)d_in[5];  // [L,B,H]
  const float* c0  = (const float*)d_in[6];  // [L,B,H]
  float* out = (float*)d_out;  // [T,B,H] ++ [L,B,H] ++ [L,B,H]

  // workspace layout (fast path is a superset of the fused path)
  char* ws = (char*)d_ws;
  int* bar = (int*)ws;  // grid barrier counter @ 0, 256B reserved
  const size_t nW = (size_t)LL * GG * HH;
  size_t off = 256;
  __bf16* wih_b = (__bf16*)(ws + off); off += nW * sizeof(__bf16);
  __bf16* whh_b = (__bf16*)(ws + off); off += nW * sizeof(__bf16);
  __bf16* hbuf  = (__bf16*)(ws + off); off += (size_t)2 * BB * HH * sizeof(__bf16);
  float*  cbuf  = (float*)(ws + off);  off += (size_t)BB * HH * sizeof(float);
  __bf16* xbuf  = (__bf16*)(ws + off); off += (size_t)TT * BB * HH * sizeof(__bf16);
  const size_t need_fused = off;
  __bf16* gx    = (__bf16*)(ws + off); off += (size_t)CT * GG * BB * sizeof(__bf16);
  const size_t need_fast = off;

  cvt_weights<<<2048, 256, 0, stream>>>(wih, whh, wih_b, whh_b, (int)nW);

  float* hT_base = out + (size_t)TT * BB * HH;
  float* cT_base = hT_base + (size_t)LL * BB * HH;

  const bool fast = (ws_size >= need_fast);
  (void)need_fused;

  if (fast) {
    for (int l = 0; l < LL; ++l) {
      const __bf16* wihl = wih_b + (size_t)l * GG * HH;
      const __bf16* whhl = whh_b + (size_t)l * GG * HH;
      const float*  bihl = bih + (size_t)l * GG;
      const float*  bhhl = bhh + (size_t)l * GG;
      float* hT = hT_base + (size_t)l * BB * HH;
      float* cT = cT_base + (size_t)l * BB * HH;
      for (int c = 0; c < TT / CT; ++c) {
        const int t0 = c * CT;
        // parallel x-side GEMM for this chunk (whole machine)
        if (l == 0)
          gates_gemm<true><<<8 * CT, 256, 0, stream>>>(x, nullptr, wihl, gx,
                                                       t0, CT);
        else
          gates_gemm<false><<<8 * CT, 256, 0, stream>>>(nullptr, xbuf, wihl,
                                                        gx, t0, CT);
        // sequential scan over the chunk (8 persistent blocks)
        hipMemsetAsync(bar, 0, sizeof(int), stream);
        if (l == 0)
          lstm_seq<0><<<NWG, 256, 0, stream>>>(
              gx, whhl, bihl, bhhl,
              (c == 0) ? h0 + (size_t)l * BB * HH : nullptr,
              (c == 0) ? c0 + (size_t)l * BB * HH : cbuf, cbuf,
              hbuf, xbuf, nullptr, hT, cT, t0, CT, bar);
        else
          lstm_seq<1><<<NWG, 256, 0, stream>>>(
              gx, whhl, bihl, bhhl,
              (c == 0) ? h0 + (size_t)l * BB * HH : nullptr,
              (c == 0) ? c0 + (size_t)l * BB * HH : cbuf, cbuf,
              hbuf, nullptr, out, hT, cT, t0, CT, bar);
      }
    }
  } else {
    // fused fallback (smaller workspace)
    hipMemsetAsync(bar, 0, sizeof(int), stream);
    lstm_fused<0><<<NWG, 256, 0, stream>>>(
        x, nullptr, wih_b, whh_b, bih, bhh, h0, c0,
        hbuf, xbuf, nullptr, hT_base, cT_base, bar);
    hipMemsetAsync(bar, 0, sizeof(int), stream);
    lstm_fused<1><<<NWG, 256, 0, stream>>>(
        nullptr, xbuf,
        wih_b + (size_t)GG * HH, whh_b + (size_t)GG * HH,
        bih + GG, bhh + GG,
        h0 + (size_t)BB * HH, c0 + (size_t)BB * HH,
        hbuf, nullptr, out,
        hT_base + (size_t)BB * HH, cT_base + (size_t)BB * HH, bar);
  }
}
